// Dynamic_Block_30958124269968
// MI455X (gfx1250) — matrix operand order repaired and checked by reading, not compiled, not run on hardware
//
#include <hip/hip_runtime.h>

// ---------------------------------------------------------------------------
// Types for CDNA5 WMMA f16 path
// ---------------------------------------------------------------------------
typedef _Float16 h8   __attribute__((ext_vector_type(8)));
typedef _Float16 v16h __attribute__((ext_vector_type(16)));
typedef float    v8f  __attribute__((ext_vector_type(8)));

#define HW   25600      // 160*160
#define NPIX 204800     // 8*160*160

// ---------------------------------------------------------------------------
// Fragment helpers (wave32; contiguous K order, same for A and B)
// A (16x32 f16, M x K): lane&15 = M; lane>>4 selects K group of 16
// contiguous K values at kbase + half*16.
// ---------------------------------------------------------------------------
__device__ __forceinline__ v16h frag_a(const _Float16* w, int ldk, int mbase,
                                       int kbase, int lane) {
    int m    = mbase + (lane & 15);
    int half_ = lane >> 4;
    const _Float16* r0 = w + (size_t)m * ldk + kbase + half_ * 16;
    const _Float16* r1 = w + (size_t)m * ldk + kbase + half_ * 16 + 8;
    h8 lo = *(const h8*)r0;
    h8 hi = *(const h8*)r1;
    v16h a;
#pragma unroll
    for (int i = 0; i < 8; ++i) { a[i] = lo[i]; a[i + 8] = hi[i]; }
    return a;
}

// B (32x16 f16, K x N): lane&15 = N (pixel); lane>>4 selects K group of 16
// contiguous K values.  `act` points at pixel 0 of the tile, row stride ldc.
__device__ __forceinline__ v16h frag_b(const _Float16* act, int ldc, int kbase,
                                       int lane) {
    const _Float16* p = act + (size_t)(lane & 15) * ldc + kbase + (lane >> 4) * 16;
    h8 lo = *(const h8*)p;
    h8 hi = *(const h8*)(p + 8);
    v16h b;
#pragma unroll
    for (int i = 0; i < 8; ++i) { b[i] = lo[i]; b[i + 8] = hi[i]; }
    return b;
}

__device__ __forceinline__ v8f wmma32(v16h a, v16h b, v8f c) {
    // v_wmma_f32_16x16x32_f16, f32 accumulate
    return __builtin_amdgcn_wmma_f32_16x16x32_f16(false, a, false, b,
                                                  (short)0, c, false, false);
}

__device__ __forceinline__ float gelu_erf(float x) {
    return 0.5f * x * (1.0f + erff(x * 0.70710678118654752f));
}

// D (16x16 f32): lane&15 = N (pixel), rows M = (lane>>4)*8 + v  (contiguous oc)
__device__ __forceinline__ void store_d_f16(v8f c, _Float16* out, int ldc,
                                            int ocbase, int lane, int do_gelu) {
    _Float16* p = out + (size_t)(lane & 15) * ldc + ocbase + (lane >> 4) * 8;
    h8 h;
#pragma unroll
    for (int v = 0; v < 8; ++v) {
        float x = c[v];
        if (do_gelu) x = gelu_erf(x);
        h[v] = (_Float16)x;
    }
    *(h8*)p = h;
}

// ---------------------------------------------------------------------------
// Prep kernels
// ---------------------------------------------------------------------------
__global__ void k_cvt_f16(const float* __restrict__ in, _Float16* __restrict__ out, int n) {
    int i = blockIdx.x * 256 + threadIdx.x;
    if (i < n) out[i] = (_Float16)in[i];
}

// w_im2 [oc][ic][ky][kx] fp32 -> w2h [tap][oc][ic] f16
__global__ void k_cvt_wim2(const float* __restrict__ w, _Float16* __restrict__ out) {
    int i = blockIdx.x * 256 + threadIdx.x;   // 64*64*9
    int oc = i / 576; int rem = i % 576;
    int ic = rem / 9; int tap = rem % 9;
    out[tap * 4096 + oc * 64 + ic] = (_Float16)w[i];
}

// Fea NCHW fp32 -> cat[p][64+c] f16
__global__ void k_cvt_fea(const float* __restrict__ Fea, _Float16* __restrict__ cat_) {
    int p = blockIdx.x * 256 + threadIdx.x;       // < NPIX
    int b = p / HW, rem = p % HW;
    _Float16* dst = cat_ + (size_t)p * 128 + 64;
    const float* src = Fea + (size_t)b * 64 * HW + rem;
#pragma unroll 8
    for (int c = 0; c < 64; ++c) dst[c] = (_Float16)src[(size_t)c * HW];
}

// conv1 3->64 (3x3, pad1) + GELU, store f16 into padded [b][162][162][64]
__global__ void __launch_bounds__(256) k_conv1(const float* __restrict__ im,
                                               const float* __restrict__ w1,
                                               _Float16* __restrict__ P1) {
    __shared__ float sw[64 * 27];
    for (int i = threadIdx.x; i < 64 * 27; i += 256) sw[i] = w1[i];
    __syncthreads();
    int p = blockIdx.x * 256 + threadIdx.x;       // < NPIX
    int b = p / HW, rem = p % HW;
    int h = rem / 160, w = rem % 160;
    float patch[27];
#pragma unroll
    for (int c = 0; c < 3; ++c)
#pragma unroll
        for (int dy = 0; dy < 3; ++dy)
#pragma unroll
            for (int dx = 0; dx < 3; ++dx) {
                int hh = h + dy - 1, ww = w + dx - 1;
                patch[c * 9 + dy * 3 + dx] =
                    (hh >= 0 && hh < 160 && ww >= 0 && ww < 160)
                        ? im[((size_t)(b * 3 + c) * 160 + hh) * 160 + ww] : 0.f;
            }
    _Float16* dst = P1 + ((size_t)(b * 162 + h + 1) * 162 + (w + 1)) * 64;
    for (int oc = 0; oc < 64; ++oc) {
        float acc = 0.f;
#pragma unroll
        for (int t = 0; t < 27; ++t) acc += sw[oc * 27 + t] * patch[t];
        dst[oc] = (_Float16)gelu_erf(acc);
    }
}

// ---------------------------------------------------------------------------
// conv2 64->64 (3x3) as implicit GEMM via WMMA.
// Each wave: 32oc x 32px macrotile (2x2 tiles, 4 accumulators).
// 9 taps x (K=64) -> 18 k-steps, 4 WMMA per step.
// ---------------------------------------------------------------------------
__global__ void __launch_bounds__(256) k_conv2(const _Float16* __restrict__ P1,
                                               const _Float16* __restrict__ w2h,
                                               _Float16* __restrict__ cat_) {
    int lane = threadIdx.x & 31, wv = threadIdx.x >> 5;
    int tile = blockIdx.x * 8 + wv;               // 12800 macrotiles
    int mg = tile & 1; int t2 = tile >> 1;        // m-group: oc 0..31 / 32..63
    int wt = t2 % 5;  int t3 = t2 / 5;            // 32-px column group
    int h = t3 % 160; int b = t3 / 160;
    const _Float16* base = P1 + ((size_t)(b * 162 + h + 1) * 162 + (wt * 32 + 1)) * 64;
    v8f c00 = {}, c01 = {}, c10 = {}, c11 = {};
#pragma unroll
    for (int tap = 0; tap < 9; ++tap) {
        int dy = tap / 3 - 1, dx = tap % 3 - 1;
        const _Float16* bptr = base + (ptrdiff_t)(dy * 162 + dx) * 64;
        const _Float16* aptr = w2h + tap * 4096;
#pragma unroll
        for (int k = 0; k < 64; k += 32) {
            v16h a0 = frag_a(aptr, 64, mg * 32, k, lane);
            v16h a1 = frag_a(aptr, 64, mg * 32 + 16, k, lane);
            v16h b0 = frag_b(bptr, 64, k, lane);
            v16h b1 = frag_b(bptr + (size_t)16 * 64, 64, k, lane);
            c00 = wmma32(a0, b0, c00);
            c01 = wmma32(a0, b1, c01);
            c10 = wmma32(a1, b0, c10);
            c11 = wmma32(a1, b1, c11);
        }
    }
    int p = (b * 160 + h) * 160 + wt * 32;
    store_d_f16(c00, cat_ + (size_t)p * 128, 128, mg * 32, lane, 0);
    store_d_f16(c10, cat_ + (size_t)p * 128, 128, mg * 32 + 16, lane, 0);
    store_d_f16(c01, cat_ + (size_t)(p + 16) * 128, 128, mg * 32, lane, 0);
    store_d_f16(c11, cat_ + (size_t)(p + 16) * 128, 128, mg * 32 + 16, lane, 0);
}

// ---------------------------------------------------------------------------
// Generic 1x1-conv GEMM (f16 in/out, optional GELU).  act row stride == K.
// Each wave: 32oc x 32px macrotile (4 accumulators), K-loop step 32.
// M must be a multiple of 32 (res1: 64, f1: 256).
// ---------------------------------------------------------------------------
__global__ void __launch_bounds__(256) k_gemm(const _Float16* __restrict__ act,
                                              const _Float16* __restrict__ w,
                                              _Float16* __restrict__ out,
                                              int K, int M, int do_gelu) {
    int lane = threadIdx.x & 31, wv = threadIdx.x >> 5;
    int tile = blockIdx.x * 8 + wv;
    int mtiles = M >> 5;                          // macro (32-row) tiles
    if (tile >= mtiles * (NPIX / 32)) return;
    int mt = tile % mtiles, nt = tile / mtiles;
    const _Float16* arow0 = act + (size_t)nt * 32 * K;
    const _Float16* arow1 = arow0 + (size_t)16 * K;
    v8f c00 = {}, c01 = {}, c10 = {}, c11 = {};
    for (int k = 0; k < K; k += 32) {
        v16h a0 = frag_a(w, K, mt * 32, k, lane);
        v16h a1 = frag_a(w, K, mt * 32 + 16, k, lane);
        v16h b0 = frag_b(arow0, K, k, lane);
        v16h b1 = frag_b(arow1, K, k, lane);
        c00 = wmma32(a0, b0, c00);
        c01 = wmma32(a0, b1, c01);
        c10 = wmma32(a1, b0, c10);
        c11 = wmma32(a1, b1, c11);
    }
    _Float16* orow0 = out + (size_t)nt * 32 * M;
    _Float16* orow1 = orow0 + (size_t)16 * M;
    store_d_f16(c00, orow0, M, mt * 32, lane, do_gelu);
    store_d_f16(c10, orow0, M, mt * 32 + 16, lane, do_gelu);
    store_d_f16(c01, orow1, M, mt * 32, lane, do_gelu);
    store_d_f16(c11, orow1, M, mt * 32 + 16, lane, do_gelu);
}

// ---------------------------------------------------------------------------
// Fused final: f2 GEMM (+bias) & res2 GEMM -> LDS, softmax over 25 taps,
// 5x5 patch contraction, residual add, pixel-shuffle x4 store.
// One block = 16 pixels of one row.
// ---------------------------------------------------------------------------
__global__ void __launch_bounds__(256) k_final(const _Float16* __restrict__ tf,
                                               const _Float16* __restrict__ tres,
                                               const _Float16* __restrict__ wf2,
                                               const _Float16* __restrict__ wres2,
                                               const float* __restrict__ bias,
                                               const float* __restrict__ im,
                                               float* __restrict__ out) {
    __shared__ float s_fx[16 * 400];
    __shared__ float s_res[16 * 48];
    __shared__ float s_patch[3 * 25 * 16];
    int lane = threadIdx.x & 31, wv = threadIdx.x >> 5;
    int blk = blockIdx.x;                         // 12800 = 8*160*10
    int wt = blk % 10; int t = blk / 10;
    int h = t % 160; int b = t / 160;
    int p0 = (b * 160 + h) * 160 + wt * 16;
    const _Float16* tfrow = tf + (size_t)p0 * 256;
    const _Float16* trrow = tres + (size_t)p0 * 64;

    for (int job = wv; job < 28; job += 8) {
        v8f c = {};
        int px = lane & 15;
        if (job < 25) {                            // filter logits, K=256
            for (int k = 0; k < 256; k += 32) {
                v16h a  = frag_a(wf2, 256, job * 16, k, lane);
                v16h bb = frag_b(tfrow, 256, k, lane);
                c = wmma32(a, bb, c);
            }
            int oc0 = job * 16 + (lane >> 4) * 8;
#pragma unroll
            for (int v = 0; v < 8; ++v)
                s_fx[px * 400 + oc0 + v] = c[v] + bias[oc0 + v];
        } else {                                   // residual, K=64
            int mt = job - 25;
#pragma unroll
            for (int k = 0; k < 64; k += 32) {
                v16h a  = frag_a(wres2, 64, mt * 16, k, lane);
                v16h bb = frag_b(trrow, 64, k, lane);
                c = wmma32(a, bb, c);
            }
            int oc0 = mt * 16 + (lane >> 4) * 8;
#pragma unroll
            for (int v = 0; v < 8; ++v)
                s_res[px * 48 + oc0 + v] = c[v];
        }
    }

    // stage 5x5 image patches: s_patch[(c*25+f)*16 + px]
    for (int i = threadIdx.x; i < 1200; i += 256) {
        int cch = i / 400; int rem = i % 400;
        int f = rem >> 4; int px = rem & 15;
        int fy = f / 5, fx = f % 5;
        int hh = h + fy - 2, ww = wt * 16 + px + fx - 2;
        float v = 0.f;
        if (hh >= 0 && hh < 160 && ww >= 0 && ww < 160)
            v = im[((size_t)(b * 3 + cch) * 160 + hh) * 160 + ww];
        s_patch[i] = v;
    }
    __syncthreads();

    int px = threadIdx.x >> 4, r = threadIdx.x & 15;
    float m = -1e30f;
#pragma unroll
    for (int f = 0; f < 25; ++f) m = fmaxf(m, s_fx[px * 400 + f * 16 + r]);
    float e[25]; float sum = 0.f;
#pragma unroll
    for (int f = 0; f < 25; ++f) {
        float ev = __expf(s_fx[px * 400 + f * 16 + r] - m);
        e[f] = ev; sum += ev;
    }
    float inv = 1.f / sum;
    float a0 = 0.f, a1 = 0.f, a2 = 0.f;
#pragma unroll
    for (int f = 0; f < 25; ++f) {
        float wgt = e[f];
        a0 += s_patch[(0 * 25 + f) * 16 + px] * wgt;
        a1 += s_patch[(1 * 25 + f) * 16 + px] * wgt;
        a2 += s_patch[(2 * 25 + f) * 16 + px] * wgt;
    }
    int rr = r >> 2, rc = r & 3;
    size_t yo = (size_t)(h * 4 + rr) * 640 + (size_t)(wt * 16 + px) * 4 + rc;
    out[(size_t)(b * 3 + 0) * 409600 + yo] = a0 * inv + s_res[px * 48 + 0 + r];
    out[(size_t)(b * 3 + 1) * 409600 + yo] = a1 * inv + s_res[px * 48 + 16 + r];
    out[(size_t)(b * 3 + 2) * 409600 + yo] = a2 * inv + s_res[px * 48 + 32 + r];
}

// ---------------------------------------------------------------------------
// Host launcher
// ---------------------------------------------------------------------------
static const size_t SZ_CAT = (size_t)NPIX * 128 * 2;         // 52,428,800
static const size_t SZ_P1  = (size_t)8 * 162 * 162 * 64 * 2; // 26,873,856
static const size_t SZ_TF  = (size_t)NPIX * 256 * 2;         // 104,857,600

extern "C" void kernel_launch(void* const* d_in, const int* in_sizes, int n_in,
                              void* d_out, int out_size, void* d_ws, size_t ws_size,
                              hipStream_t stream) {
    const float* Fea    = (const float*)d_in[0];
    const float* im     = (const float*)d_in[1];
    const float* w_im1  = (const float*)d_in[2];
    const float* w_im2  = (const float*)d_in[3];
    const float* w_res1 = (const float*)d_in[4];
    const float* w_res2 = (const float*)d_in[5];
    const float* w_f1   = (const float*)d_in[6];
    const float* w_f2   = (const float*)d_in[7];
    const float* b_f2   = (const float*)d_in[8];
    float* out = (float*)d_out;
    (void)in_sizes; (void)n_in; (void)out_size; (void)ws_size;

    char* ws = (char*)d_ws;
    _Float16* cat_ = (_Float16*)(ws);
    _Float16* P1   = (_Float16*)(ws + SZ_CAT);
    _Float16* tres = P1;                          // reuse after conv2 consumed P1
    _Float16* tf   = (_Float16*)(ws + SZ_CAT + SZ_P1);
    char* wb = ws + SZ_CAT + SZ_P1 + SZ_TF;
    _Float16* w2h = (_Float16*)(wb);              //  73,728 B
    _Float16* wr1 = (_Float16*)(wb + 73728);      //  16,384 B
    _Float16* wr2 = (_Float16*)(wb + 90112);      //   6,144 B
    _Float16* wf1 = (_Float16*)(wb + 96256);      //  65,536 B
    _Float16* wf2 = (_Float16*)(wb + 161792);     // 204,800 B

    hipMemsetAsync(P1, 0, SZ_P1, stream);         // zero border for conv2 taps

    k_cvt_f16<<<(8192 + 255) / 256, 256, 0, stream>>>(w_res1, wr1, 8192);
    k_cvt_f16<<<(3072 + 255) / 256, 256, 0, stream>>>(w_res2, wr2, 3072);
    k_cvt_f16<<<(32768 + 255) / 256, 256, 0, stream>>>(w_f1, wf1, 32768);
    k_cvt_f16<<<(102400 + 255) / 256, 256, 0, stream>>>(w_f2, wf2, 102400);
    k_cvt_wim2<<<144, 256, 0, stream>>>(w_im2, w2h);

    k_conv1<<<NPIX / 256, 256, 0, stream>>>(im, w_im1, P1);
    k_cvt_fea<<<NPIX / 256, 256, 0, stream>>>(Fea, cat_);
    k_conv2<<<1600, 256, 0, stream>>>(P1, w2h, cat_);

    k_gemm<<<1600, 256, 0, stream>>>(cat_, wr1, tres, 128, 64, 1);    // res1+GELU
    k_gemm<<<6400, 256, 0, stream>>>(cat_, wf1, tf, 128, 256, 1);     // f1+GELU

    k_final<<<12800, 256, 0, stream>>>(tf, tres, wf2, wr2, b_f2, im, out);
}
